// SpatialAttentionLinear_77446850281995
// MI455X (gfx1250) — matrix operand order repaired and checked by reading, not compiled, not run on hardware
//
#include <hip/hip_runtime.h>

// Problem constants (fixed by the reference)
#define BB   16
#define HH   32
#define WW   32
#define CC   256
#define NN   1024            // H*W tokens
#define NTOK (BB * NN)       // 16384

typedef __attribute__((ext_vector_type(16))) __bf16          v16bf;
typedef __attribute__((ext_vector_type(8)))  __bf16          v8bf;
typedef __attribute__((ext_vector_type(16))) unsigned short  v16u;
typedef __attribute__((ext_vector_type(8)))  unsigned short  v8u;
typedef __attribute__((ext_vector_type(16))) float           v16f;
typedef __attribute__((ext_vector_type(8)))  float           v8f;

union F16 { v16bf h; v16u u; };
union F8  { v8bf  h; v8u  u; };

__device__ __forceinline__ unsigned short f2bf(float f) {
    // scalar round-to-nearest-even f32 -> bf16 (used only in scalar paths)
    unsigned int u = __float_as_uint(f);
    u += 0x7FFFu + ((u >> 16) & 1u);
    return (unsigned short)(u >> 16);
}

__device__ __forceinline__ v8f zero8() {
    v8f z = {0.f, 0.f, 0.f, 0.f, 0.f, 0.f, 0.f, 0.f};
    return z;
}

__device__ __forceinline__ v8f wmma_bf16(const F16& a, const F16& b, v8f c) {
    return __builtin_amdgcn_wmma_f32_16x16x32_bf16(
        /*neg_a=*/false, a.h, /*neg_b=*/false, b.h,
        /*c_mod=*/(short)0, c, /*reuse_a=*/false, /*reuse_b=*/false);
}

#define SHUF16(lo, hi) __builtin_shufflevector(lo, hi, 0,1,2,3,4,5,6,7,8,9,10,11,12,13,14,15)

// CDNA5 async global->LDS copy (ASYNCcnt-tracked); VDST = per-lane LDS byte
// address, VADDR = 64-bit global address. Inline asm per bridge doc (the
// clang builtins differ in arity across toolchains).
__device__ __forceinline__ void async_load_b128(unsigned lds_off, const void* gptr) {
    asm volatile("global_load_async_to_lds_b128 %0, %1, off"
                 :: "v"(lds_off), "v"(gptr) : "memory");
}
__device__ __forceinline__ void wait_async0() {
    asm volatile("s_wait_asynccnt 0x0" ::: "memory");
}

// reduce across the 16-lane half-wave (xor offsets < 16 stay inside the half)
__device__ __forceinline__ float hmax16(float v) {
    #pragma unroll
    for (int off = 8; off >= 1; off >>= 1) v = fmaxf(v, __shfl_xor(v, off, 32));
    return v;
}
__device__ __forceinline__ float hsum16(float v) {
    #pragma unroll
    for (int off = 8; off >= 1; off >>= 1) v += __shfl_xor(v, off, 32);
    return v;
}

// ---------------------------------------------------------------------------
// Kernel 0: WpT[h][w][n] = bf16(Wp[h][n][w]) so kernel-3 A-fragments read the
// K axis (n) contiguously. 1M elements, coalesced writes.
// ---------------------------------------------------------------------------
__global__ __launch_bounds__(256) void wp_transpose_kernel(
    const float* __restrict__ Wp, unsigned short* __restrict__ WpT)
{
    const int idx = blockIdx.x * 256 + threadIdx.x;   // [0, HH*WW*NN)
    const int n  = idx & (NN - 1);
    const int hw = idx >> 10;                          // h*WW + w
    const int w  = hw & (WW - 1);
    const int h  = hw >> 5;
    WpT[idx] = f2bf(Wp[((size_t)h * NN + n) * WW + w]);
}

// ---------------------------------------------------------------------------
// Kernel 1: fused q/k/v projection. q = x @ Wq^T + bq (likewise k, v).
// q,k stored row-major [token][c]; v stored transposed vT[b][c][n] via a
// single packed b128 store per lane (C/D layout: a lane holds 8 consecutive
// rows of one column). 49152 tiles; 8 waves/block -> 6144 blocks.
// ---------------------------------------------------------------------------
__global__ __launch_bounds__(256) void qkv_proj_kernel(
    const float* __restrict__ x,
    const float* __restrict__ Wq, const float* __restrict__ bq,
    const float* __restrict__ Wk, const float* __restrict__ bk,
    const float* __restrict__ Wv, const float* __restrict__ bv,
    unsigned short* __restrict__ qB, unsigned short* __restrict__ kB,
    unsigned short* __restrict__ vT)
{
    const int lane = threadIdx.x & 31;
    const int wave = threadIdx.x >> 5;
    const int tile = blockIdx.x * 8 + wave;        // [0, 49152)
    const int ntile = tile & 15;
    const int p     = (tile >> 4) % 3;
    const int mtile = tile / 48;

    const float* Wm = (p == 0) ? Wq : ((p == 1) ? Wk : Wv);
    const float* bm = (p == 0) ? bq : ((p == 1) ? bk : bv);

    const int lm   = lane & 15;
    const int half = lane >> 4;
    const int tokenA = mtile * 16 + lm;            // A-matrix row (token)
    const int dcol   = ntile * 16 + lm;            // B-matrix column (out chan)

    v8f acc = zero8();
    for (int kb = 0; kb < CC; kb += 32) {
        // A (16x32 bf16): elems 0..15 -> K=half*16+e
        const float* xr = x + (size_t)tokenA * CC + kb;
        v8f xlo = *(const v8f*)(xr + half * 16);
        v8f xhi = *(const v8f*)(xr + half * 16 + 8);
        F16 a; a.h = __builtin_convertvector(SHUF16(xlo, xhi), v16bf);
        // B (32x16): lane = column n, elems = 16 consecutive K per half.
        // B[k=c][n=d] = W[d][c] -> contiguous 64B read of row d.
        const float* wr = Wm + (size_t)dcol * CC + kb + half * 16;
        v16f wf = *(const v16f*)wr;
        F16 b; b.h = __builtin_convertvector(wf, v16bf);
        acc = wmma_bf16(a, b, acc);
    }
    const v8f accb = acc + bm[dcol];               // splat bias over rows
    F8 pk; pk.h = __builtin_convertvector(accb, v8bf);

    if (p == 2) {
        // vT[c][n]: lane's 8 rows are consecutive n -> one 16B store
        *(v8u*)(vT + (size_t)dcol * NN + mtile * 16 + 8 * half) = pk.u;
    } else {
        unsigned short* outB = (p == 0) ? qB : kB;
        #pragma unroll
        for (int r = 0; r < 8; ++r)
            outB[(size_t)(mtile * 16 + r + 8 * half) * CC + dcol] = pk.u[r];
    }
}

// ---------------------------------------------------------------------------
// Kernel 2: streaming-softmax attention. All 8 waves of a block share one
// batch, so each 32-key K chunk (16KB) and V chunk (16KB) is staged ONCE per
// block into LDS with CDNA5 async global->LDS copies (ASYNCcnt), then all
// waves build WMMA fragments from LDS. Row padding (K:+8 -> 132 dwords/row,
// V:+8 -> 20 dwords/row) spreads the 16 rows a half-wave touches across
// disjoint 4-bank groups of the 64 LDS banks.
// Tiles: 16 batches * 64 qtiles = 1024; 8 waves/block -> 128 blocks.
// ---------------------------------------------------------------------------
__global__ __launch_bounds__(256) void attn_kernel(
    const unsigned short* __restrict__ qB,
    const unsigned short* __restrict__ kB,
    const unsigned short* __restrict__ vT,
    unsigned short* __restrict__ oT)
{
    __shared__ unsigned short ldsK[32][264];       // 32 keys x 256 c (+8 pad)
    __shared__ unsigned short ldsV[256][40];       // 256 c x 32 n  (+8 pad)
    __shared__ unsigned short ldsP[8][16][32];     // per-wave P tiles

    const int lane = threadIdx.x & 31;
    const int wave = threadIdx.x >> 5;
    const int tile = blockIdx.x * 8 + wave;        // [0, 1024)
    const int batch = tile >> 6;                   // same for all waves in blk
    const int qtile = tile & 63;
    const int lm   = lane & 15;
    const int half = lane >> 4;

    const unsigned short* qb = qB + (size_t)batch * NN * CC;
    const unsigned short* kb = kB + (size_t)batch * NN * CC;
    const unsigned short* vb = vT + (size_t)batch * CC * NN;

    // Preload the 16x256 Q tile as 8 A-fragments (b128 loads)
    F16 qf[8];
    const int qrow = qtile * 16 + lm;
    #pragma unroll
    for (int kk = 0; kk < 8; ++kk) {
        const unsigned short* qr = qb + (size_t)qrow * CC + kk * 32;
        v8u lo = *(const v8u*)(qr + half * 16);
        v8u hi = *(const v8u*)(qr + half * 16 + 8);
        qf[kk].u = SHUF16(lo, hi);
    }

    v8f o[16];
    #pragma unroll
    for (int f = 0; f < 16; ++f) o[f] = zero8();
    float rmax[8], rsum[8];
    #pragma unroll
    for (int r = 0; r < 8; ++r) { rmax[r] = -3.0e38f; rsum[r] = 0.f; }

    for (int jb = 0; jb < NN; jb += 32) {
        // Cooperative async fill: 256 threads x 4 x 16B each for K and V.
        // (trailing barrier of the previous iteration guarantees the LDS
        //  buffers are no longer being read)
        #pragma unroll
        for (int i = 0; i < 4; ++i) {
            const int g  = i * 256 + threadIdx.x;       // [0, 1024)
            const int kr = g >> 5, kc = (g & 31) * 8;   // 32 rows x 256 elems
            async_load_b128((unsigned)(uintptr_t)&ldsK[kr][kc],
                            kb + (size_t)(jb + kr) * CC + kc);
            const int vr = g >> 2, vc = (g & 3) * 8;    // 256 rows x 32 elems
            async_load_b128((unsigned)(uintptr_t)&ldsV[vr][vc],
                            vb + (size_t)vr * NN + jb + vc);
        }
        wait_async0();          // own async copies landed in LDS
        __syncthreads();        // everyone's copies visible to all waves

        // scores for two 16-key tiles (local key rows lm and 16+lm)
        v8f s0 = zero8(), s1 = zero8();
        #pragma unroll
        for (int kk = 0; kk < 8; ++kk) {
            F16 b0, b1;
            {
                v8u lo = *(const v8u*)&ldsK[lm][kk * 32 + half * 16];
                v8u hi = *(const v8u*)&ldsK[lm][kk * 32 + half * 16 + 8];
                b0.u = SHUF16(lo, hi);
            }
            {
                v8u lo = *(const v8u*)&ldsK[16 + lm][kk * 32 + half * 16];
                v8u hi = *(const v8u*)&ldsK[16 + lm][kk * 32 + half * 16 + 8];
                b1.u = SHUF16(lo, hi);
            }
            s0 = wmma_bf16(qf[kk], b0, s0);
            s1 = wmma_bf16(qf[kk], b1, s1);
        }
        // online softmax update; row M = r + 8*half lives across 16 lanes
        #pragma unroll
        for (int r = 0; r < 8; ++r) {
            float tmax = hmax16(fmaxf(s0[r], s1[r]));
            float mnew = fmaxf(rmax[r], tmax);
            float scale = __expf(rmax[r] - mnew);
            rmax[r] = mnew;
            float p0 = __expf(s0[r] - mnew);
            float p1 = __expf(s1[r] - mnew);
            rsum[r] = rsum[r] * scale + hsum16(p0 + p1);
            #pragma unroll
            for (int f = 0; f < 16; ++f) o[f][r] *= scale;
            const int row = r + 8 * half;
            ldsP[wave][row][lm]      = f2bf(p0);
            ldsP[wave][row][16 + lm] = f2bf(p1);
        }
        // C-layout -> A-layout transpose via this wave's private LDS tile
        // (same-wave DS ops are in-order)
        F16 pa;
        {
            v8u lo = *(const v8u*)&ldsP[wave][lm][half * 16];
            v8u hi = *(const v8u*)&ldsP[wave][lm][half * 16 + 8];
            pa.u = SHUF16(lo, hi);
        }
        // O += P * V over 16 c-tiles; ldsV rows are c, columns are local n
        #pragma unroll
        for (int ct = 0; ct < 16; ++ct) {
            F16 vf;
            v8u lo = *(const v8u*)&ldsV[ct * 16 + lm][half * 16];
            v8u hi = *(const v8u*)&ldsV[ct * 16 + lm][half * 16 + 8];
            vf.u = SHUF16(lo, hi);
            o[ct] = wmma_bf16(pa, vf, o[ct]);
        }
        __syncthreads();        // all waves done reading before next fill
    }

    // normalize rows, pack bf16, store transposed oT[c][n] (b128 stores)
    unsigned short* ob = oT + (size_t)batch * CC * NN;
    v8f invv;
    #pragma unroll
    for (int r = 0; r < 8; ++r) invv[r] = 1.0f / rsum[r];
    const int nrow0 = qtile * 16 + 8 * half;       // 8 consecutive n per lane
    #pragma unroll
    for (int f = 0; f < 16; ++f) {
        F8 pk; pk.h = __builtin_convertvector(o[f] * invv, v8bf);
        *(v8u*)(ob + (size_t)(f * 16 + lm) * NN + nrow0) = pk.u;
    }
}

// ---------------------------------------------------------------------------
// Kernel 3: z[b,h,w,c] = sum_n qkv[b,n,c] * Wp[h,n,w]
// Per (b,h): GEMM M=32(w), N=256(c), K=1024(n). A[w,n]=WpT[h][w][n] (bf16,
// contiguous K), B[n,c]=oT[b][c][n] (bf16, contiguous K). All b128 loads.
// Tiles: 16*32*(2 mtiles)*(16 ntiles) = 16384; 8 waves/block -> 2048 blocks.
// ---------------------------------------------------------------------------
__global__ __launch_bounds__(256) void zproj_kernel(
    const unsigned short* __restrict__ oT,
    const unsigned short* __restrict__ WpT,
    float* __restrict__ z)
{
    const int lane = threadIdx.x & 31;
    const int wave = threadIdx.x >> 5;
    const int tile = blockIdx.x * 8 + wave;        // [0, 16384)
    const int nt = tile & 15;
    const int mt = (tile >> 4) & 1;
    const int h  = (tile >> 5) & 31;
    const int b  = tile >> 10;
    const int lm   = lane & 15;
    const int half = lane >> 4;
    const int wcol = mt * 16 + lm;                 // A row (w)
    const int ccol = nt * 16 + lm;                 // B column (c)

    const unsigned short* wr = WpT + (size_t)(h * WW + wcol) * NN;
    const unsigned short* ob = oT + ((size_t)b * CC + ccol) * NN;

    v8f acc = zero8();
    for (int kbase = 0; kbase < NN; kbase += 32) {
        F16 a, bf;
        v8u lo = *(const v8u*)(wr + kbase + half * 16);
        v8u hi = *(const v8u*)(wr + kbase + half * 16 + 8);
        a.u  = SHUF16(lo, hi);
        bf.u = *(const v16u*)(ob + kbase + half * 16);
        acc = wmma_bf16(a, bf, acc);
    }
    #pragma unroll
    for (int r = 0; r < 8; ++r) {
        const int w = mt * 16 + r + 8 * half;
        z[(((size_t)b * HH + h) * WW + w) * CC + ccol] = acc[r];
    }
}

// ---------------------------------------------------------------------------
extern "C" void kernel_launch(void* const* d_in, const int* in_sizes, int n_in,
                              void* d_out, int out_size, void* d_ws, size_t ws_size,
                              hipStream_t stream) {
    const float* x  = (const float*)d_in[0];
    const float* Wq = (const float*)d_in[1];
    const float* bq = (const float*)d_in[2];
    const float* Wk = (const float*)d_in[3];
    const float* bk = (const float*)d_in[4];
    const float* Wv = (const float*)d_in[5];
    const float* bv = (const float*)d_in[6];
    const float* Wp = (const float*)d_in[7];
    float* z = (float*)d_out;

    // Workspace (bf16): q, k row-major; vT, oT transposed [b][c][n]; WpT.
    // 4 * 8 MB + 2 MB = 34 MB, all resident in the 192 MB L2.
    unsigned short* qB  = (unsigned short*)d_ws;
    unsigned short* kB  = qB + (size_t)NTOK * CC;
    unsigned short* vT  = kB + (size_t)NTOK * CC;
    unsigned short* oT  = vT + (size_t)NTOK * CC;
    unsigned short* WpT = oT + (size_t)NTOK * CC;

    wp_transpose_kernel<<<4096, 256, 0, stream>>>(Wp, WpT);
    qkv_proj_kernel<<<6144, 256, 0, stream>>>(x, Wq, bq, Wk, bk, Wv, bv, qB, kB, vT);
    attn_kernel   <<<128,  256, 0, stream>>>(qB, kB, vT, oT);
    zproj_kernel  <<<2048, 256, 0, stream>>>(oT, WpT, z);
}
